// SE3Net_15702400434540
// MI455X (gfx1250) — matrix operand order repaired and checked by reading, not compiled, not run on hardware
//
#include <hip/hip_runtime.h>
#include <stdint.h>

// ---------------------------------------------------------------------------
// Types / helpers
// ---------------------------------------------------------------------------
typedef __attribute__((ext_vector_type(16))) __bf16 bf16x16;
typedef __attribute__((ext_vector_type(8)))  float  f32x8;

union Frag { bf16x16 v; uint4 q[2]; };

__device__ __forceinline__ unsigned short f2b(float f) {
  union { float f; unsigned u; } x; x.f = f;
  unsigned r = x.u + 0x7fffu + ((x.u >> 16) & 1u);   // round-to-nearest-even
  return (unsigned short)(r >> 16);
}
__device__ __forceinline__ float b2f(unsigned short h) {
  union { unsigned u; float f; } x; x.u = ((unsigned)h) << 16;
  return x.f;
}
__device__ __forceinline__ float ldf(const float* p, int i)          { return p[i]; }
__device__ __forceinline__ float ldf(const unsigned short* p, int i) { return b2f(p[i]); }

// ---------------------------------------------------------------------------
// Weight transform: fp32 W[O][C][125] -> bf16 Wb[Mp][125*Cpad]
//   k = tap*Cpad + ic ; zero for ic>=C or m>=O
// ---------------------------------------------------------------------------
__global__ void se3_convw(const float* __restrict__ W, unsigned short* __restrict__ Wb,
                          int O, int C, int Cpad, int total) {
  int idx = blockIdx.x * 256 + threadIdx.x;
  if (idx >= total) return;
  const int Ktot = 125 * Cpad;
  const int m   = idx / Ktot;
  int r         = idx - m * Ktot;
  const int tap = r / Cpad;
  const int ic  = r - tap * Cpad;
  float v = (m < O && ic < C) ? W[((size_t)m * C + ic) * 125 + tap] : 0.0f;
  Wb[idx] = f2b(v);
}

// ---------------------------------------------------------------------------
// NHWC implicit-GEMM conv3d (5^3, pad 4), bf16 WMMA, f32 accumulate.
//   X  : [S_in^3][CPAD]      bf16 (channel-last, zero padded channels)
//   Wb : [Mp][125*CPAD]      bf16
//   Y  : [S_out^3][Mp]       bf16
// Block: 256 threads = 8 waves, 2(M) x 4(N); tile 32 x 128. Each wave keeps
// two 16x16 f32 accumulators; per 32-wide K chunk: 2 global A loads, 4 global
// B loads (contiguous 16B each), 2 v_wmma_f32_16x16x32_bf16. No LDS, no
// barriers — B fragments are direct contiguous channel reads.
// ---------------------------------------------------------------------------
template <int CPAD>
__global__ __launch_bounds__(256) void se3_conv_wmma(
    const unsigned short* __restrict__ X,
    const unsigned short* __restrict__ Wb,
    unsigned short* __restrict__ Y,
    int S_in, int S_out, int Mp) {
  const int S2 = S_out * S_out;
  const int S3 = S2 * S_out;
  constexpr int KTOT = 125 * CPAD;

  const int t     = threadIdx.x;
  const int wave  = t >> 5;
  const int lane  = t & 31;
  const int laneM = lane & 15;
  const int hi    = lane >> 4;
  const int wm    = wave >> 2;     // 0..1  M sub-tile
  const int wn    = wave & 3;      // 0..3  N sub-tile

  // two output columns (voxels) handled by this lane's fragments
  const int col0 = blockIdx.x * 128 + wn * 32 + laneM;
  const int col1 = col0 + 16;
  const bool v0 = col0 < S3, v1 = col1 < S3;
  const int cc0 = v0 ? col0 : 0;
  const int cc1 = v1 ? col1 : 0;
  const int oz0 = cc0 / S2; int r0 = cc0 - oz0 * S2;
  const int oy0 = r0 / S_out;  const int ox0 = r0 - oy0 * S_out;
  const int oz1 = cc1 / S2; int r1 = cc1 - oz1 * S2;
  const int oy1 = r1 / S_out;  const int ox1 = r1 - oy1 * S_out;
  const int bz0 = oz0 - 4, by0 = oy0 - 4, bx0 = ox0 - 4;
  const int bz1 = oz1 - 4, by1 = oy1 - 4, bx1 = ox1 - 4;

  const uint4* __restrict__ Arow =
      (const uint4*)(Wb + (size_t)(blockIdx.y * 32 + wm * 16 + laneM) * KTOT);

  f32x8 c0 = {}, c1 = {};
  const uint4 zq = make_uint4(0u, 0u, 0u, 0u);

  int tap = 0;
  for (int kz = 0; kz < 5; ++kz) {
    const int iz0 = bz0 + kz, iz1 = bz1 + kz;
    const bool vz0 = v0 && iz0 >= 0 && iz0 < S_in;
    const bool vz1 = v1 && iz1 >= 0 && iz1 < S_in;
    for (int ky = 0; ky < 5; ++ky) {
      const int iy0 = by0 + ky, iy1 = by1 + ky;
      const bool vy0 = vz0 && iy0 >= 0 && iy0 < S_in;
      const bool vy1 = vz1 && iy1 >= 0 && iy1 < S_in;
      const int rb0 = (iz0 * S_in + iy0) * S_in;
      const int rb1 = (iz1 * S_in + iy1) * S_in;
#pragma unroll
      for (int kx = 0; kx < 5; ++kx, ++tap) {
        const int ix0 = bx0 + kx, ix1 = bx1 + kx;
        const bool val0 = vy0 && ix0 >= 0 && ix0 < S_in;
        const bool val1 = vy1 && ix1 >= 0 && ix1 < S_in;
        const unsigned short* p0 = X + (size_t)(rb0 + ix0) * CPAD + hi * 16;
        const unsigned short* p1 = X + (size_t)(rb1 + ix1) * CPAD + hi * 16;
#pragma unroll
        for (int ch = 0; ch < CPAD; ch += 32) {
          Frag a, b0, b1;
          const uint4* ap = Arow + ((tap * CPAD + ch) >> 3) + hi * 2;
          a.q[0] = ap[0];          // K = k0 + hi*16     .. +7
          a.q[1] = ap[1];          // K = k0 + hi*16 + 8 .. +7
          if (val0) {
            const uint4* p = (const uint4*)(p0 + ch);
            b0.q[0] = p[0]; b0.q[1] = p[1];
          } else { b0.q[0] = zq; b0.q[1] = zq; }
          if (val1) {
            const uint4* p = (const uint4*)(p1 + ch);
            b1.q[0] = p[0]; b1.q[1] = p[1];
          } else { b1.q[0] = zq; b1.q[1] = zq; }
          c0 = __builtin_amdgcn_wmma_f32_16x16x32_bf16(false, a.v, false, b0.v,
                                                       (short)0, c0, false, false);
          c1 = __builtin_amdgcn_wmma_f32_16x16x32_bf16(false, a.v, false, b1.v,
                                                       (short)0, c1, false, false);
        }
      }
    }
  }

  // epilogue: 8 accumulator rows = 8 consecutive output channels -> one 16B store
  const int ocBase = blockIdx.y * 32 + wm * 16 + hi * 8;
  if (v0) {
    union { unsigned short h[8]; uint4 q; } o;
#pragma unroll
    for (int v = 0; v < 8; ++v) o.h[v] = f2b(c0[v]);
    *(uint4*)(Y + (size_t)col0 * Mp + ocBase) = o.q;
  }
  if (v1) {
    union { unsigned short h[8]; uint4 q; } o;
#pragma unroll
    for (int v = 0; v < 8; ++v) o.h[v] = f2b(c1[v]);
    *(uint4*)(Y + (size_t)col1 * Mp + ocBase) = o.q;
  }
}

// ---------------------------------------------------------------------------
// Depthwise Gaussian low-pass (scale=2): 5^3, pad 2, stride s, channel-last.
// g = normalized exp(-r^2/1.5), r=-2..2. Channels >= C are zero-filled.
// ---------------------------------------------------------------------------
template <typename TIN>
__global__ void se3_lowpass(const TIN* __restrict__ X, unsigned short* __restrict__ Y,
                            int C, int CinS, int CoutS,
                            int S_in, int S_out, int stride, int total) {
  const float G[5] = {0.03208120f, 0.23705652f, 0.46172310f, 0.23705652f, 0.03208120f};
  int idx = blockIdx.x * 256 + threadIdx.x;
  if (idx >= total) return;
  const int s = idx / CoutS;
  const int c = idx - s * CoutS;
  if (c >= C) { Y[idx] = 0; return; }
  const int S2 = S_out * S_out;
  const int oz = s / S2; int rem = s - oz * S2;
  const int oy = rem / S_out;
  const int ox = rem - oy * S_out;
  float acc = 0.0f;
  for (int kz = 0; kz < 5; ++kz) {
    const int iz = oz * stride + kz - 2;
    if (iz < 0 || iz >= S_in) continue;
    for (int ky = 0; ky < 5; ++ky) {
      const int iy = oy * stride + ky - 2;
      if (iy < 0 || iy >= S_in) continue;
      const float gzy = G[kz] * G[ky];
      const int rowb = (iz * S_in + iy) * S_in;
      for (int kx = 0; kx < 5; ++kx) {
        const int ix = ox * stride + kx - 2;
        if (ix < 0 || ix >= S_in) continue;
        acc += gzy * G[kx] * ldf(X, (rowb + ix) * CinS + c);
      }
    }
  }
  Y[idx] = f2b(acc);
}

// ---------------------------------------------------------------------------
// Gated nonlinearity, channel-last. Channels >= Cout are zero-filled pads.
// ---------------------------------------------------------------------------
__global__ void se3_gated(const unsigned short* __restrict__ X, unsigned short* __restrict__ Y,
                          int CinS, int CoutS, int Cout, int n_scalar, int n_gate,
                          int m0, int d0, int m1, int d1, int m2, int d2, int total) {
  int idx = blockIdx.x * 256 + threadIdx.x;
  if (idx >= total) return;
  const int s = idx / CoutS;
  const int c = idx - s * CoutS;
  if (c >= Cout) { Y[idx] = 0; return; }
  const unsigned short* base = X + (size_t)s * CinS;
  float out;
  if (c < n_scalar) {
    out = fmaxf(b2f(base[c]), 0.0f);
  } else {
    int f = c - n_scalar;
    int gprev = 0, fprev = 0, gch = 0, fch = 0;
    const int ms[3] = {m0, m1, m2};
    const int ds[3] = {d0, d1, d2};
#pragma unroll
    for (int i = 0; i < 3; ++i) {
      const int sz = ms[i] * ds[i];
      if (f < sz) {
        gch = n_scalar + gprev + f / ds[i];
        fch = n_scalar + n_gate + fprev + f;
        break;
      }
      f -= sz; gprev += ms[i]; fprev += sz;
    }
    const float g = 1.0f / (1.0f + expf(-b2f(base[gch])));
    out = g * b2f(base[fch]);
  }
  Y[idx] = f2b(out);
}

// ---------------------------------------------------------------------------
// relu + spatial mean, channel-last input (one block per channel)
// ---------------------------------------------------------------------------
__global__ void se3_reduce(const unsigned short* __restrict__ X, float* __restrict__ pooled,
                           int Cstride, int S3, float inv) {
  __shared__ float sm[256];
  const int c = blockIdx.x;
  float acc = 0.0f;
  for (int i = threadIdx.x; i < S3; i += 256)
    acc += fmaxf(b2f(X[(size_t)i * Cstride + c]), 0.0f);
  sm[threadIdx.x] = acc;
  __syncthreads();
  for (int s = 128; s > 0; s >>= 1) {
    if (threadIdx.x < s) sm[threadIdx.x] += sm[threadIdx.x + s];
    __syncthreads();
  }
  if (threadIdx.x == 0) pooled[c] = sm[0] * inv;
}

// ---------------------------------------------------------------------------
// final linear: out[b,j] = pooled[b,:] . Wl[j,:] + bl[j]   (16x10)
// ---------------------------------------------------------------------------
__global__ void se3_linear(const float* __restrict__ pooled, const float* __restrict__ Wl,
                           const float* __restrict__ bl, float* __restrict__ out) {
  const int t = threadIdx.x;
  if (t >= 160) return;
  const int b = t / 10, j = t - b * 10;
  float acc = bl[j];
  for (int k = 0; k < 64; ++k) acc += pooled[b * 64 + k] * Wl[j * 64 + k];
  out[b * 10 + j] = acc;
}

// ---------------------------------------------------------------------------
// Launch
// ---------------------------------------------------------------------------
extern "C" void kernel_launch(void* const* d_in, const int* in_sizes, int n_in,
                              void* d_out, int out_size, void* d_ws, size_t ws_size,
                              hipStream_t stream) {
  const float* x  = (const float*)d_in[0];
  const float* W1 = (const float*)d_in[1];
  const float* W2 = (const float*)d_in[2];
  const float* W3 = (const float*)d_in[3];
  const float* W4 = (const float*)d_in[4];
  const float* Wl = (const float*)d_in[5];
  const float* bl = (const float*)d_in[6];
  float* out = (float*)d_out;

  char* base = (char*)d_ws;
  size_t off = 0;
  auto alloc = [&](size_t bytes) -> char* {
    char* p = base + off;
    off = (off + bytes + 255) & ~(size_t)255;
    return p;
  };
  auto cdiv = [](int a, int b) { return (a + b - 1) / b; };

  constexpr int B = 16;
  // geometry
  constexpr int S1i3 = 32768;                       // 32^3
  constexpr int S1o3 = 46656;                       // 36^3
  constexpr int S2o3 = 64000, S2p3 = 8000;          // 40^3 -> 20^3
  constexpr int S3o3 = 13824, S3p3 = 1728;          // 24^3 -> 12^3
  constexpr int S4o3 = 4096;                        // 16^3
  // GEMM shapes
  constexpr int M1p = 32, M2p = 96, M3p = 96, M4p = 64;
  constexpr int CP1 = 32, CP2 = 32, CP3 = 96, CP4 = 96;   // input channel pads

  unsigned short* W1b  = (unsigned short*)alloc((size_t)M1p * 125 * CP1 * 2);
  unsigned short* W2b  = (unsigned short*)alloc((size_t)M2p * 125 * CP2 * 2);
  unsigned short* W3b  = (unsigned short*)alloc((size_t)M3p * 125 * CP3 * 2);
  unsigned short* W4b  = (unsigned short*)alloc((size_t)M4p * 125 * CP4 * 2);
  unsigned short* lp1b = (unsigned short*)alloc((size_t)S1i3 * CP1 * 2);   // per-batch
  unsigned short* g1b  = (unsigned short*)alloc((size_t)S1o3 * CP2 * 2);
  unsigned short* g2b  = (unsigned short*)alloc((size_t)S2p3 * CP3 * 2);
  unsigned short* g3b  = (unsigned short*)alloc((size_t)S3p3 * CP4 * 2);
  unsigned short* tmpC = (unsigned short*)alloc((size_t)S2o3 * 96 * 2);    // conv out (max)
  unsigned short* tmpL = (unsigned short*)alloc((size_t)S2p3 * 96 * 2);    // low-pass out (max)
  float* pooled = (float*)alloc((size_t)B * 64 * 4);

  // weights -> bf16 tap-major/channel-minor GEMM layout
  se3_convw<<<cdiv(M1p * 125 * CP1, 256), 256, 0, stream>>>(W1, W1b, 30, 1,  CP1, M1p * 125 * CP1);
  se3_convw<<<cdiv(M2p * 125 * CP2, 256), 256, 0, stream>>>(W2, W2b, 88, 25, CP2, M2p * 125 * CP2);
  se3_convw<<<cdiv(M3p * 125 * CP3, 256), 256, 0, stream>>>(W3, W3b, 96, 72, CP3, M3p * 125 * CP3);
  se3_convw<<<cdiv(M4p * 125 * CP4, 256), 256, 0, stream>>>(W4, W4b, 64, 80, CP4, M4p * 125 * CP4);

  for (int b = 0; b < B; ++b) {
    // ---- input low-pass (stride 1), fp32 -> bf16 NHWC[32] ----
    se3_lowpass<float><<<cdiv(S1i3 * CP1, 256), 256, 0, stream>>>(
        x + (size_t)b * S1i3, lp1b, 1, 1, CP1, 32, 32, 1, S1i3 * CP1);

    // ---- layer 1: conv 1->30 @36^3, gated -> 25ch (pad 32) ----
    se3_conv_wmma<CP1><<<dim3(cdiv(S1o3, 128), M1p / 32), 256, 0, stream>>>(
        lp1b, W1b, tmpC, 32, 36, M1p);
    se3_gated<<<cdiv(S1o3 * CP2, 256), 256, 0, stream>>>(
        tmpC, g1b, M1p, CP2, 25, 2, 5, 2, 3, 2, 5, 1, 7, S1o3 * CP2);

    // ---- layer 2: conv 25->88 @40^3, lp/2 -> 20^3, gated -> 72ch (pad 96) ----
    se3_conv_wmma<CP2><<<dim3(cdiv(S2o3, 128), M2p / 32), 256, 0, stream>>>(
        g1b, W2b, tmpC, 36, 40, M2p);
    se3_lowpass<unsigned short><<<cdiv(S2p3 * 96, 256), 256, 0, stream>>>(
        tmpC, tmpL, 88, M2p, 96, 40, 20, 2, S2p3 * 96);
    se3_gated<<<cdiv(S2p3 * CP3, 256), 256, 0, stream>>>(
        tmpL, g2b, 96, CP3, 72, 8, 16, 8, 3, 8, 5, 0, 1, S2p3 * CP3);

    // ---- layer 3: conv 72->96 @24^3, lp/2 -> 12^3, gated -> 80ch (pad 96) ----
    se3_conv_wmma<CP3><<<dim3(cdiv(S3o3, 128), M3p / 32), 256, 0, stream>>>(
        g2b, W3b, tmpC, 20, 24, M3p);
    se3_lowpass<unsigned short><<<cdiv(S3p3 * 96, 256), 256, 0, stream>>>(
        tmpC, tmpL, 96, M3p, 96, 24, 12, 2, S3p3 * 96);
    se3_gated<<<cdiv(S3p3 * CP4, 256), 256, 0, stream>>>(
        tmpL, g3b, 96, CP4, 80, 16, 16, 8, 3, 8, 5, 0, 1, S3p3 * CP4);

    // ---- layer 4: conv 80->64 @16^3, relu + mean -> pooled[b] ----
    se3_conv_wmma<CP4><<<dim3(cdiv(S4o3, 128), M4p / 32), 256, 0, stream>>>(
        g3b, W4b, tmpC, 12, 16, M4p);
    se3_reduce<<<64, 256, 0, stream>>>(tmpC, pooled + b * 64, M4p, S4o3, 1.0f / S4o3);
  }

  se3_linear<<<1, 256, 0, stream>>>(pooled, Wl, bl, out);
}
